// EPRFormerBlock_2336462209426
// MI455X (gfx1250) — matrix operand order repaired and checked by reading, not compiled, not run on hardware
//
#include <hip/hip_runtime.h>
#include <hip/hip_bf16.h>

// ---------------------------------------------------------------------------
// EPRFormer block for gfx1250 (MI455X).  All dense contractions use
// v_wmma_f32_16x16x32_bf16 (wave32 WMMA).  Fragment layouts used here (k contiguous in
// A and B alike):  A(16x32 bf16): lane l holds row m=l&15, halves h -> k=h+16*(l>=16)
//          B(32x16 bf16): lane l holds col n=l&15, halves h -> k=h+16*(l>=16)
//          C/D(16x16 f32): lane l, vgpr v -> m=v+8*(l>=16), n=l&15
// conv1x1 stages activations in LDS transposed ([pixel][channel], padded rows)
// so B-fragments are two contiguous ds_load_b128 per WMMA.
// ---------------------------------------------------------------------------

typedef __attribute__((ext_vector_type(16))) __bf16 bf16x16;
typedef __attribute__((ext_vector_type(8)))  float  f32x8;

#define NB     8
#define DIMC   128
#define HH     96
#define WW2    96
#define HWP    9216            // 96*96
#define NWX    12
#define NWIN   144             // 12*12 windows
#define WSZ    8
#define TOKW   64              // tokens per window
#define TOPKN  4
#define KVLEN  256             // TOPK*64
#define KVP    264             // KVLEN + 8 pad (keeps 16B row alignment)
#define NHEAD  4
#define HDIM   32
#define HID    256
#define BN_RS  0.9999950000374996f
#define NT_HW  576             // HW/16

__device__ __forceinline__ f32x8 wmma_bf16(bf16x16 a, bf16x16 b, f32x8 c) {
  return __builtin_amdgcn_wmma_f32_16x16x32_bf16(false, a, false, b, (short)0, c,
                                                 false, false);
}
__device__ __forceinline__ int akp(int h, int g16) { return h + (g16 << 4); }

// 32B LDS vector load of a bf16x16 fragment (2 x ds_load_b128)
__device__ __forceinline__ bf16x16 lds_load_bf16x16(const __bf16* p) {
  union { uint4 u[2]; bf16x16 v; } t;
  t.u[0] = ((const uint4*)p)[0];
  t.u[1] = ((const uint4*)p)[1];
  return t.v;
}
// packed store of 8 f32 accumulators as 8 contiguous bf16 (1 x ds_store_b128)
__device__ __forceinline__ void lds_store_bf16x8(__bf16* p, const f32x8 acc) {
  union { uint4 u; __bf16 b[8]; } t;
#pragma unroll
  for (int v = 0; v < 8; ++v) t.b[v] = (__bf16)acc[v];
  *((uint4*)p) = t.u;
}

// ---------------------------------------------------------------------------
// window means for x and prompt
__global__ void win_mean_kernel(const float* __restrict__ x,
                                const float* __restrict__ prompt,
                                float* __restrict__ xd, float* __restrict__ pd) {
  int i = blockIdx.x * blockDim.x + threadIdx.x;
  int total = NB * NWIN * DIMC;
  if (i >= 2 * total) return;
  const float* src = (i < total) ? x : prompt;
  float* dst       = (i < total) ? xd : pd;
  int j = (i < total) ? i : i - total;
  int c = j % DIMC; int n = (j / DIMC) % NWIN; int b = j / (DIMC * NWIN);
  int wy = n / NWX, wx = n % NWX;
  const float* p = src + ((size_t)b * DIMC + c) * HWP + wy * WSZ * WW2 + wx * WSZ;
  float s = 0.f;
  for (int ty = 0; ty < WSZ; ++ty)
    for (int tx = 0; tx < WSZ; ++tx) s += p[ty * WW2 + tx];
  dst[j] = s * (1.f / 64.f);
}

// ---------------------------------------------------------------------------
// scores + top-4 (descending, first-index-on-ties, matching lax.top_k)
__global__ void topk_kernel(const float* __restrict__ xd,
                            const float* __restrict__ pd, int* __restrict__ idx) {
  __shared__ float sx[DIMC];
  __shared__ float sc[NWIN];
  int b = blockIdx.x / NWIN, n = blockIdx.x % NWIN;
  int t = threadIdx.x;                       // 128 threads
  sx[t] = xd[(b * NWIN + n) * DIMC + t];
  __syncthreads();
  for (int m = t; m < NWIN; m += 128) {
    float s = 0.f;
    const float* pr = pd + (b * NWIN + m) * DIMC;
    for (int c = 0; c < DIMC; ++c) s += sx[c] * pr[c];
    sc[m] = s;                               // scale doesn't change ranking
  }
  __syncthreads();
  if (t == 0) {
    for (int j = 0; j < TOPKN; ++j) {
      float best = -1e30f; int bi = 0;
      for (int m = 0; m < NWIN; ++m) if (sc[m] > best) { best = sc[m]; bi = m; }
      idx[(b * NWIN + n) * TOPKN + j] = bi;
      sc[bi] = -1e30f;
    }
  }
}

// ---------------------------------------------------------------------------
// fused prompt-routing attention: one block (8 waves) per (batch, window)
// LDS: sQ[64][128], sK[256][128] (row-major), sV transposed [128][KVP],
//      sO[64][128], per-wave score rows, sInv, sIdx.
__global__ void __launch_bounds__(256) prompt_attn_kernel(
    const float* __restrict__ x, const float* __restrict__ prompt,
    const int* __restrict__ idx,
    const float* __restrict__ Wq, const float* __restrict__ Wk,
    const float* __restrict__ Wv, const float* __restrict__ Wp,
    float* __restrict__ Y) {
  extern __shared__ char smem[];
  __bf16* sQ  = (__bf16*)smem;                 // 64 x 128
  __bf16* sK  = sQ + TOKW * DIMC;              // 256 x 128
  __bf16* sVT = sK + KVLEN * DIMC;             // 128 x KVP   (transposed V)
  __bf16* sO  = sVT + DIMC * KVP;              // 64 x 128
  float*  sScore = (float*)(sO + TOKW * DIMC); // 8 waves x 16 x 256
  float*  sInv   = sScore + 8 * 16 * 256;      // 8 x 16
  int*    sIdx   = (int*)(sInv + 8 * 16);      // 4

  int b  = blockIdx.x / NWIN;
  int nw = blockIdx.x % NWIN;
  int wy = nw / NWX, wx = nw % NWX;
  int lane = threadIdx.x & 31;
  int wv   = threadIdx.x >> 5;
  int m15  = lane & 15;
  int g16  = lane >> 4;
  int kb16 = g16 << 4;

  if (threadIdx.x < TOPKN)
    sIdx[threadIdx.x] = idx[(b * NWIN + nw) * TOPKN + threadIdx.x];
  __syncthreads();

  const float* xb = x + (size_t)b * DIMC * HWP;
  const float* pb = prompt + (size_t)b * DIMC * HWP;

  // ---- Q = xw @ Wq^T : 4 m-tiles x 8 n-tiles; wave owns n-tile = wv ----
  {
    int n = wv;
#pragma unroll
    for (int mt = 0; mt < 4; ++mt) {
      int t = mt * 16 + m15;
      int pix = (wy * WSZ + (t >> 3)) * WW2 + wx * WSZ + (t & 7);
      f32x8 acc = {};
#pragma unroll
      for (int kc = 0; kc < 4; ++kc) {
        bf16x16 af, bfr;
#pragma unroll
        for (int h = 0; h < 16; ++h)
          af[h] = (__bf16)xb[(size_t)(kc * 32 + akp(h, g16)) * HWP + pix];
        const float* wp = Wq + (size_t)(n * 16 + m15) * DIMC + kc * 32 + kb16;
#pragma unroll
        for (int h = 0; h < 16; ++h) bfr[h] = (__bf16)wp[h];
        acc = wmma_bf16(af, bfr, acc);
      }
#pragma unroll
      for (int v = 0; v < 8; ++v)
        sQ[(mt * 16 + v + 8 * g16) * DIMC + n * 16 + m15] = (__bf16)acc[v];
    }
  }

  // ---- K,V = sel @ Wk^T / Wv^T : 16 m-tiles x 8 n-tiles; share A frags ----
#pragma unroll
  for (int mi = 0; mi < 2; ++mi) {
    int mt = wv * 2 + mi;
    int t  = mt * 16 + m15;                 // kv token 0..255
    int sw = t >> 6, ti = t & 63;
    int pidx = sIdx[sw];
    int ppix = ((pidx / NWX) * WSZ + (ti >> 3)) * WW2 + (pidx % NWX) * WSZ + (ti & 7);
    bf16x16 af[4];
#pragma unroll
    for (int kc = 0; kc < 4; ++kc)
#pragma unroll
      for (int h = 0; h < 16; ++h)
        af[kc][h] = (__bf16)pb[(size_t)(kc * 32 + akp(h, g16)) * HWP + ppix];
#pragma unroll
    for (int nt = 0; nt < 8; ++nt) {
      f32x8 aK = {}, aV = {};
#pragma unroll
      for (int kc = 0; kc < 4; ++kc) {
        bf16x16 bk, bv;
        const float* wkp = Wk + (size_t)(nt * 16 + m15) * DIMC + kc * 32 + kb16;
        const float* wvp = Wv + (size_t)(nt * 16 + m15) * DIMC + kc * 32 + kb16;
#pragma unroll
        for (int h = 0; h < 16; ++h) { bk[h] = (__bf16)wkp[h]; bv[h] = (__bf16)wvp[h]; }
        aK = wmma_bf16(af[kc], bk, aK);
        aV = wmma_bf16(af[kc], bv, aV);
      }
#pragma unroll
      for (int v = 0; v < 8; ++v)
        sK[(mt * 16 + v + 8 * g16) * DIMC + nt * 16 + m15] = (__bf16)aK[v];
      // V transposed: row = channel (nt*16+m15), 8 consecutive tokens packed
      lds_store_bf16x8(&sVT[(nt * 16 + m15) * KVP + mt * 16 + 8 * g16], aV);
    }
  }
  __syncthreads();

  // ---- attention: waves 0-3 head (hp*2), waves 4-7 head (hp*2+1) ----
  float* sS = sScore + wv * 16 * 256;
#pragma unroll
  for (int hp = 0; hp < 2; ++hp) {
    int h  = hp * 2 + (wv >> 2);
    int qm = wv & 3;                         // q-row tile
    bf16x16 aq;
#pragma unroll
    for (int hh = 0; hh < 16; ++hh)
      aq[hh] = sQ[(qm * 16 + m15) * DIMC + h * HDIM + akp(hh, g16)];
#pragma unroll
    for (int nt = 0; nt < 16; ++nt) {        // K=32 -> single WMMA per tile
      bf16x16 bk = lds_load_bf16x16(&sK[(nt * 16 + m15) * DIMC + h * HDIM + kb16]);
      f32x8 acc = {};
      acc = wmma_bf16(aq, bk, acc);
#pragma unroll
      for (int v = 0; v < 8; ++v)
        sS[(v + 8 * g16) * 256 + nt * 16 + m15] = acc[v] * 0.17677669529663687f;
    }
    __syncthreads();
    if (lane < 16) {                          // per-row softmax (no WMMA here)
      float* row = sS + lane * 256;
      float mx = -1e30f;
      for (int j = 0; j < 256; ++j) mx = fmaxf(mx, row[j]);
      float sum = 0.f;
      for (int j = 0; j < 256; ++j) { float e = __expf(row[j] - mx); row[j] = e; sum += e; }
      sInv[wv * 16 + lane] = 1.f / sum;
    }
    __syncthreads();
    float rinv = sInv[wv * 16 + m15];
#pragma unroll
    for (int et = 0; et < 2; ++et) {
      f32x8 acc = {};
#pragma unroll
      for (int kc = 0; kc < 8; ++kc) {
        bf16x16 ap;
#pragma unroll
        for (int hh = 0; hh < 16; ++hh)
          ap[hh] = (__bf16)(sS[m15 * 256 + kc * 32 + akp(hh, g16)] * rinv);
        bf16x16 bv = lds_load_bf16x16(
            &sVT[(h * HDIM + et * 16 + m15) * KVP + kc * 32 + kb16]);
        acc = wmma_bf16(ap, bv, acc);
      }
#pragma unroll
      for (int v = 0; v < 8; ++v)
        sO[(qm * 16 + v + 8 * g16) * DIMC + h * HDIM + et * 16 + m15] = (__bf16)acc[v];
    }
    __syncthreads();
  }

  // ---- proj: y = o @ Wp^T, scatter back to NCHW ----
  {
    int n = wv;
#pragma unroll
    for (int mt = 0; mt < 4; ++mt) {
      f32x8 acc = {};
#pragma unroll
      for (int kc = 0; kc < 4; ++kc) {
        bf16x16 ao, bp;
#pragma unroll
        for (int h = 0; h < 16; ++h)
          ao[h] = sO[(mt * 16 + m15) * DIMC + kc * 32 + akp(h, g16)];
        const float* wp = Wp + (size_t)(n * 16 + m15) * DIMC + kc * 32 + kb16;
#pragma unroll
        for (int h = 0; h < 16; ++h) bp[h] = (__bf16)wp[h];
        acc = wmma_bf16(ao, bp, acc);
      }
#pragma unroll
      for (int v = 0; v < 8; ++v) {
        int t = mt * 16 + v + 8 * g16;
        int pix = (wy * WSZ + (t >> 3)) * WW2 + wx * WSZ + (t & 7);
        Y[((size_t)b * DIMC + n * 16 + m15) * HWP + pix] = acc[v];
      }
    }
  }
}

// ---------------------------------------------------------------------------
// generic 1x1-conv GEMM (WMMA) with fused epilogues, LDS-staged activations.
//   out[b][o][p] = f( sum_k W[o][k] * in(k)[p] )
//   block = (batch, 16-pixel tile); wave w handles m-tiles w, w+8, ...
//   activation tile staged transposed [pixel][channel] (row pad 8) so each
//   B-fragment is 2 contiguous ds_load_b128.
//   in(k): Xa for k<Csplit, Xb for k>=Csplit; optional input affine (BN fold).
//   epilogue: optional BN (g,b), act (1=SiLU), then
//     gmul!=0 : out = gadd + gmul * sigmoid(v)
//     res !=0 : out = res + v
template <int Cc, int Mt>
__global__ void __launch_bounds__(256) conv1x1_kernel(
    const float* __restrict__ Xa, const float* __restrict__ Xb, int Csplit,
    const float* __restrict__ W, float* __restrict__ Y,
    const float* __restrict__ bng, const float* __restrict__ bnb,
    const float* __restrict__ res,
    const float* __restrict__ ing, const float* __restrict__ inb, int act,
    const float* __restrict__ gmul, const float* __restrict__ gadd) {
  constexpr int O = Mt * 16;
  constexpr int CP = Cc + 8;                   // padded row (16B aligned)
  __shared__ __bf16 sXT[16 * CP];              // [pixel][channel]
  int lane = threadIdx.x & 31;
  int wv   = threadIdx.x >> 5;
  int b  = blockIdx.x / NT_HW;
  int nt = blockIdx.x % NT_HW;
  int p0 = nt * 16;
  int m15 = lane & 15, g16 = lane >> 4, kb16 = g16 << 4;
  int Ca = Xb ? Csplit : Cc;
  const float* xa = Xa + (size_t)b * Ca * HWP;
  const float* xbp = Xb ? Xb + (size_t)b * (Cc - Csplit) * HWP : nullptr;

  // stage activation tile: thread loads float4 (4 pixels of one channel),
  // writes 4 bf16 into transposed LDS rows.
#pragma unroll
  for (int e = threadIdx.x * 4; e < Cc * 16; e += 256 * 4) {
    int k = e >> 4, n4 = e & 15;
    const float* src = xa; int kk = k;
    if (xbp && k >= Csplit) { src = xbp; kk = k - Csplit; }
    const float4 v4 = *(const float4*)(src + (size_t)kk * HWP + p0 + n4);
    float vs[4] = {v4.x, v4.y, v4.z, v4.w};
    if (ing) {
      float sc = ing[k] * BN_RS, bi = inb[k];
#pragma unroll
      for (int q = 0; q < 4; ++q) vs[q] = vs[q] * sc + bi;
    }
#pragma unroll
    for (int q = 0; q < 4; ++q) sXT[(n4 + q) * CP + k] = (__bf16)vs[q];
  }
  __syncthreads();

#pragma unroll
  for (int mt = wv; mt < Mt; mt += 8) {
    int o0 = mt * 16;
    f32x8 acc = {};
#pragma unroll
    for (int kc = 0; kc < Cc; kc += 32) {
      bf16x16 af;
      const float* wp = W + (size_t)(o0 + m15) * Cc + kc;
#pragma unroll
      for (int h = 0; h < 16; ++h) af[h] = (__bf16)wp[akp(h, g16)];
      bf16x16 bfr = lds_load_bf16x16(&sXT[m15 * CP + kc + kb16]);
      acc = wmma_bf16(af, bfr, acc);
    }
#pragma unroll
    for (int v = 0; v < 8; ++v) {
      int oc = o0 + v + 8 * g16;
      int pix = p0 + m15;
      size_t oi = ((size_t)b * O + oc) * HWP + pix;
      float val = acc[v];
      if (bng) val = val * bng[oc] * BN_RS + bnb[oc];
      if (act == 1) val = val * (1.f / (1.f + __expf(-val)));   // SiLU
      if (gmul) {
        val = gadd[oi] + gmul[oi] * (1.f / (1.f + __expf(-val)));
      } else if (res) {
        val += res[oi];
      }
      Y[oi] = val;
    }
  }
}

// ---------------------------------------------------------------------------
// qkv = qkv + 0.5*(dw3 + dw5), C = 384
__global__ void dw_qkv_kernel(const float* __restrict__ in,
                              const float* __restrict__ w3,
                              const float* __restrict__ w5,
                              float* __restrict__ out) {
  size_t i = (size_t)blockIdx.x * blockDim.x + threadIdx.x;
  size_t total = (size_t)NB * 384 * HWP;
  if (i >= total) return;
  int px = i % WW2; int py = (i / WW2) % HH;
  int c  = (i / HWP) % 384; int b = (int)(i / ((size_t)384 * HWP));
  const float* src = in + ((size_t)b * 384 + c) * HWP;
  const float* k3 = w3 + c * 9; const float* k5 = w5 + c * 25;
  float s3 = 0.f, s5 = 0.f;
  for (int dy = -2; dy <= 2; ++dy) {
    int yy = py + dy; if (yy < 0 || yy >= HH) continue;
    for (int dx = -2; dx <= 2; ++dx) {
      int xx = px + dx; if (xx < 0 || xx >= WW2) continue;
      float v = src[yy * WW2 + xx];
      s5 += v * k5[(dy + 2) * 5 + (dx + 2)];
      if (dy >= -1 && dy <= 1 && dx >= -1 && dx <= 1)
        s3 += v * k3[(dy + 1) * 3 + (dx + 1)];
    }
  }
  out[i] = src[py * WW2 + px] + 0.5f * (s3 + s5);
}

// ctx = dw3 + dw5 + dw7, C = 256
__global__ void msca_ctx_kernel(const float* __restrict__ in,
                                const float* __restrict__ w3,
                                const float* __restrict__ w5,
                                const float* __restrict__ w7,
                                float* __restrict__ out) {
  size_t i = (size_t)blockIdx.x * blockDim.x + threadIdx.x;
  size_t total = (size_t)NB * HID * HWP;
  if (i >= total) return;
  int px = i % WW2; int py = (i / WW2) % HH;
  int c  = (i / HWP) % HID; int b = (int)(i / ((size_t)HID * HWP));
  const float* src = in + ((size_t)b * HID + c) * HWP;
  const float* k3 = w3 + c * 9; const float* k5 = w5 + c * 25; const float* k7 = w7 + c * 49;
  float s = 0.f;
  for (int dy = -3; dy <= 3; ++dy) {
    int yy = py + dy; if (yy < 0 || yy >= HH) continue;
    for (int dx = -3; dx <= 3; ++dx) {
      int xx = px + dx; if (xx < 0 || xx >= WW2) continue;
      float v = src[yy * WW2 + xx];
      s += v * k7[(dy + 3) * 7 + (dx + 3)];
      if (dy >= -2 && dy <= 2 && dx >= -2 && dx <= 2)
        s += v * k5[(dy + 2) * 5 + (dx + 2)];
      if (dy >= -1 && dy <= 1 && dx >= -1 && dx <= 1)
        s += v * k3[(dy + 1) * 3 + (dx + 1)];
    }
  }
  out[i] = s;
}

// ---------------------------------------------------------------------------
// linear attention: kv[b,h,d,e] = sum_p relu(k[p,d]) * v[p,e]   (WMMA, K=pixels)
__global__ void la_kv_kernel(const float* __restrict__ qkv, float* __restrict__ kv) {
  int lane = threadIdx.x & 31, wv = threadIdx.x >> 5;
  int b = blockIdx.x / NHEAD, h = blockIdx.x % NHEAD;
  int mt = wv >> 1, nt = wv & 1;
  int m15 = lane & 15, g16 = lane >> 4, kb16 = g16 << 4;
  const float* qb = qkv + (size_t)b * 3 * DIMC * HWP;
  const float* kc_ = qb + (size_t)(DIMC + h * HDIM + mt * 16 + m15) * HWP;
  const float* vc_ = qb + (size_t)(2 * DIMC + h * HDIM + nt * 16 + m15) * HWP;
  f32x8 acc = {};
  for (int pc = 0; pc < HWP; pc += 32) {
    bf16x16 af, bfr;
#pragma unroll
    for (int hh = 0; hh < 16; ++hh) af[hh] = (__bf16)fmaxf(kc_[pc + akp(hh, g16)], 0.f);
#pragma unroll
    for (int hh = 0; hh < 16; ++hh) bfr[hh] = (__bf16)vc_[pc + hh + kb16];
    acc = wmma_bf16(af, bfr, acc);
  }
#pragma unroll
  for (int v = 0; v < 8; ++v)
    kv[(((size_t)(b * NHEAD + h)) * HDIM + mt * 16 + v + 8 * g16) * HDIM + nt * 16 + m15] = acc[v];
}

__global__ void la_ksum_kernel(const float* __restrict__ qkv, float* __restrict__ ksum) {
  int i = blockIdx.x * blockDim.x + threadIdx.x;
  if (i >= NB * NHEAD * HDIM) return;
  int d = i % HDIM; int h = (i / HDIM) % NHEAD; int b = i / (HDIM * NHEAD);
  const float* kc = qkv + ((size_t)b * 3 * DIMC + DIMC + h * HDIM + d) * HWP;
  float s = 0.f;
  for (int p = 0; p < HWP; ++p) s += fmaxf(kc[p], 0.f);
  ksum[i] = s;
}

__global__ void la_norm_kernel(const float* __restrict__ qkv,
                               const float* __restrict__ ksum, float* __restrict__ nrm) {
  size_t i = (size_t)blockIdx.x * blockDim.x + threadIdx.x;
  if (i >= (size_t)NB * NHEAD * HWP) return;
  int p = i % HWP; int h = (int)((i / HWP) % NHEAD); int b = (int)(i / ((size_t)HWP * NHEAD));
  const float* qc = qkv + ((size_t)b * 3 * DIMC + h * HDIM) * HWP + p;
  const float* ks = ksum + (b * NHEAD + h) * HDIM;
  float s = 0.f;
  for (int d = 0; d < HDIM; ++d) s += fmaxf(qc[(size_t)d * HWP], 0.f) * ks[d];
  nrm[i] = fmaxf(s, 1e-6f);
}

// o[b, h*32+e, p] = (relu(q) @ kv)[p,e] / nrm   (one WMMA per 16x16 tile, K=32)
__global__ void __launch_bounds__(256) la_o_kernel(
    const float* __restrict__ qkv, const float* __restrict__ kv,
    const float* __restrict__ nrm, float* __restrict__ O) {
  int lane = threadIdx.x & 31, wv = threadIdx.x >> 5;
  int tile = blockIdx.x * 8 + wv;                // NB*NHEAD*576*2 tiles
  int nt2 = tile & 1; int r = tile >> 1;
  int mt = r % NT_HW; int bh = r / NT_HW;
  if (bh >= NB * NHEAD) return;
  int b = bh / NHEAD, h = bh % NHEAD;
  int m15 = lane & 15, g16 = lane >> 4, kb16 = g16 << 4;
  int p0 = mt * 16;
  const float* qc = qkv + ((size_t)b * 3 * DIMC + h * HDIM) * HWP;
  const float* kvb = kv + (size_t)bh * HDIM * HDIM;
  bf16x16 af, bfr;
#pragma unroll
  for (int hh = 0; hh < 16; ++hh)
    af[hh] = (__bf16)fmaxf(qc[(size_t)akp(hh, g16) * HWP + p0 + m15], 0.f);
#pragma unroll
  for (int hh = 0; hh < 16; ++hh)
    bfr[hh] = (__bf16)kvb[(hh + kb16) * HDIM + nt2 * 16 + m15];
  f32x8 acc = {};
  acc = wmma_bf16(af, bfr, acc);
#pragma unroll
  for (int v = 0; v < 8; ++v) {
    int pix = p0 + v + 8 * g16;
    float val = acc[v] / nrm[(size_t)bh * HWP + pix];
    O[((size_t)b * DIMC + h * HDIM + nt2 * 16 + m15) * HWP + pix] = val;
  }
}

// ---------------------------------------------------------------------------
extern "C" void kernel_launch(void* const* d_in, const int* in_sizes, int n_in,
                              void* d_out, int out_size, void* d_ws, size_t ws_size,
                              hipStream_t stream) {
  const float* x      = (const float*)d_in[0];
  const float* prompt = (const float*)d_in[1];
  const float* pr_q = (const float*)d_in[2];
  const float* pr_k = (const float*)d_in[3];
  const float* pr_v = (const float*)d_in[4];
  const float* pr_proj = (const float*)d_in[5];
  const float* pr_gate_w = (const float*)d_in[6];
  const float* pr_gate_g = (const float*)d_in[7];
  const float* pr_gate_b = (const float*)d_in[8];
  const float* la_qkv = (const float*)d_in[9];
  const float* la_dw3 = (const float*)d_in[10];
  const float* la_dw5 = (const float*)d_in[11];
  const float* la_proj_w = (const float*)d_in[12];
  const float* la_proj_g = (const float*)d_in[13];
  const float* la_proj_b = (const float*)d_in[14];
  const float* ms_norm_g = (const float*)d_in[15];
  const float* ms_norm_b = (const float*)d_in[16];
  const float* ms_exp_w = (const float*)d_in[17];
  const float* ms_exp_g = (const float*)d_in[18];
  const float* ms_exp_b = (const float*)d_in[19];
  const float* ms_dw3 = (const float*)d_in[20];
  const float* ms_dw5 = (const float*)d_in[21];
  const float* ms_dw7 = (const float*)d_in[22];
  const float* ms_attn_w = (const float*)d_in[23];
  const float* ms_proj_w = (const float*)d_in[24];
  const float* ms_proj_g = (const float*)d_in[25];
  const float* ms_proj_b = (const float*)d_in[26];
  const float* ff_w1 = (const float*)d_in[27];
  const float* ff_g1 = (const float*)d_in[28];
  const float* ff_b1 = (const float*)d_in[29];
  const float* ff_w2 = (const float*)d_in[30];
  const float* ff_g2 = (const float*)d_in[31];
  const float* ff_b2 = (const float*)d_in[32];
  float* out = (float*)d_out;

  // workspace layout (floats).  requires ~343 MB of scratch.
  float* ws = (float*)d_ws;
  float* xd   = ws;                       // 147456
  float* pd   = xd + 147456;              // 147456
  int*   idxb = (int*)(pd + 147456);      // 4608
  float* kvb  = (float*)(idxb + 4608);    // 32768
  float* ksum = kvb + 32768;              // 1024
  float* nrm  = ksum + 1024;              // 294912
  float* big  = nrm + 294912;
  const size_t S = (size_t)NB * DIMC * HWP;   // 9437184
  float* R1 = big;            // 2S: Y -> O_la -> Y2
  float* R2 = R1 + 2 * S;     // 1S: X1 -> X3
  float* R3 = R2 + S;         // 3S: QKVa -> {X2, CTX}
  float* R4 = R3 + 3 * S;     // 3S: QKVb -> {YM, H1}
  float* Ybuf = R1; float* Ola = R1; float* Y2 = R1;
  float* X1 = R2; float* X3 = R2;
  float* QKVa = R3; float* X2 = R3; float* CTX = R3 + S;
  float* QKVb = R4; float* YM = R4; float* H1 = R4;

  const int CONV_GRID = NB * NT_HW;       // one block per (batch, pixel tile)

  // ---- prompt routing ----
  win_mean_kernel<<<(2 * NB * NWIN * DIMC + 255) / 256, 256, 0, stream>>>(x, prompt, xd, pd);
  topk_kernel<<<NB * NWIN, 128, 0, stream>>>(xd, pd, idxb);
  const size_t pa_smem =
      (size_t)(TOKW * DIMC + KVLEN * DIMC + DIMC * KVP + TOKW * DIMC) * sizeof(__bf16) +
      (size_t)(8 * 16 * 256 + 8 * 16) * sizeof(float) + 4 * sizeof(int) + 16;
  prompt_attn_kernel<<<NB * NWIN, 256, pa_smem, stream>>>(x, prompt, idxb, pr_q, pr_k,
                                                          pr_v, pr_proj, Ybuf);
  // X1 = x + sigmoid(BN(Wg @ [x;y])) * y
  conv1x1_kernel<256, 8><<<CONV_GRID, 256, 0, stream>>>(
      x, Ybuf, DIMC, pr_gate_w, X1, pr_gate_g, pr_gate_b,
      nullptr, nullptr, nullptr, 0, Ybuf, x);

  // ---- linear attention ----
  conv1x1_kernel<128, 24><<<CONV_GRID, 256, 0, stream>>>(
      X1, nullptr, 0, la_qkv, QKVa, nullptr, nullptr,
      nullptr, nullptr, nullptr, 0, nullptr, nullptr);
  dw_qkv_kernel<<<(unsigned)(((size_t)NB * 384 * HWP) / 256), 256, 0, stream>>>(
      QKVa, la_dw3, la_dw5, QKVb);
  la_kv_kernel<<<NB * NHEAD, 128, 0, stream>>>(QKVb, kvb);
  la_ksum_kernel<<<(NB * NHEAD * HDIM + 255) / 256, 256, 0, stream>>>(QKVb, ksum);
  la_norm_kernel<<<(unsigned)(((size_t)NB * NHEAD * HWP) / 256), 256, 0, stream>>>(
      QKVb, ksum, nrm);
  la_o_kernel<<<NB * NHEAD * NT_HW * 2 / 8, 256, 0, stream>>>(QKVb, kvb, nrm, Ola);
  // X2 = X1 + BN(proj @ o)
  conv1x1_kernel<128, 8><<<CONV_GRID, 256, 0, stream>>>(
      Ola, nullptr, 0, la_proj_w, X2, la_proj_g, la_proj_b,
      X1, nullptr, nullptr, 0, nullptr, nullptr);

  // ---- MSCA ----
  // YM = silu(BN(exp @ BN(X2)))
  conv1x1_kernel<128, 16><<<CONV_GRID, 256, 0, stream>>>(
      X2, nullptr, 0, ms_exp_w, YM, ms_exp_g, ms_exp_b,
      nullptr, ms_norm_g, ms_norm_b, 1, nullptr, nullptr);
  msca_ctx_kernel<<<(unsigned)(((size_t)NB * HID * HWP) / 256), 256, 0, stream>>>(
      YM, ms_dw3, ms_dw5, ms_dw7, CTX);
  // Y2 = YM + CTX * sigmoid(attn_w @ CTX)
  conv1x1_kernel<256, 16><<<CONV_GRID, 256, 0, stream>>>(
      CTX, nullptr, 0, ms_attn_w, Y2, nullptr, nullptr,
      nullptr, nullptr, nullptr, 0, CTX, YM);
  // X3 = X2 + BN(proj @ Y2)
  conv1x1_kernel<256, 8><<<CONV_GRID, 256, 0, stream>>>(
      Y2, nullptr, 0, ms_proj_w, X3, ms_proj_g, ms_proj_b,
      X2, nullptr, nullptr, 0, nullptr, nullptr);

  // ---- FFN ----
  conv1x1_kernel<128, 16><<<CONV_GRID, 256, 0, stream>>>(
      X3, nullptr, 0, ff_w1, H1, ff_g1, ff_b1,
      nullptr, nullptr, nullptr, 1, nullptr, nullptr);
  conv1x1_kernel<256, 8><<<CONV_GRID, 256, 0, stream>>>(
      H1, nullptr, 0, ff_w2, out, ff_g2, ff_b2,
      X3, nullptr, nullptr, 0, nullptr, nullptr);

  (void)in_sizes; (void)n_in; (void)out_size; (void)ws_size;
}
